// MultiHeadAttention_50809463111698
// MI455X (gfx1250) — matrix operand order repaired and checked by reading, not compiled, not run on hardware
//
#include <hip/hip_runtime.h>

// ---------------------------------------------------------------- types
typedef __attribute__((ext_vector_type(16))) __bf16 v16bf;
typedef __attribute__((ext_vector_type(8)))  __bf16 v8bf;
typedef __attribute__((ext_vector_type(8)))  float  v8f;
typedef __attribute__((ext_vector_type(4)))  int    v4i;

union AFrag { v16bf v; v8bf h[2]; };

static __device__ __forceinline__ v8f wmma_bf16(v16bf a, v16bf b, v8f c) {
  return __builtin_amdgcn_wmma_f32_16x16x32_bf16(
      /*neg_a=*/false, a, /*neg_b=*/false, b,
      /*c_mod=*/(short)0, c, /*reuse_a=*/false, /*reuse_b=*/false);
}

// Problem constants (fixed by the reference)
#define SEQ_   2048
#define DM_    1024
#define NH_    16
#define DH_    64
#define MTOT_  8192   // B*S

// ------------------------------------------------ async global->LDS (CDNA5)
#if defined(__HIP_DEVICE_COMPILE__) && \
    __has_builtin(__builtin_amdgcn_global_load_async_to_lds_b128)
#define HAVE_ASYNC_LDS 1
#else
#define HAVE_ASYNC_LDS 0
#endif

// copy 16 bytes from global to LDS (per lane)
static __device__ __forceinline__ void async_cp16(const __bf16* g, __bf16* l) {
#if HAVE_ASYNC_LDS
  __builtin_amdgcn_global_load_async_to_lds_b128(
      (__attribute__((address_space(1))) v4i*)g,
      (__attribute__((address_space(3))) v4i*)l, 0, 0);
#else
  *(uint4*)l = *(const uint4*)g;
#endif
}

static __device__ __forceinline__ void wait_async_all() {
#if HAVE_ASYNC_LDS
#if __has_builtin(__builtin_amdgcn_s_wait_asynccnt)
  __builtin_amdgcn_s_wait_asynccnt(0);
#else
  asm volatile("s_wait_asynccnt 0x0" ::: "memory");
#endif
#endif
}

// ---------------------------------------------------------------- f32 -> bf16
__global__ __launch_bounds__(256) void cvt_f32_bf16(
    const float* __restrict__ in, __bf16* __restrict__ out, int n) {
  int i = (blockIdx.x * 256 + threadIdx.x) * 4;
  if (i < n) {
    float4 f = *(const float4*)(in + i);
    out[i + 0] = (__bf16)f.x;
    out[i + 1] = (__bf16)f.y;
    out[i + 2] = (__bf16)f.z;
    out[i + 3] = (__bf16)f.w;
  }
}

// ---------------------------------------------------------------- GEMM
// C[M,N] = A[M,K] @ W[N,K]^T + bias
// Block tile 128(M) x 128(N) x 32(K); 8 waves in 4(M) x 2(N); each wave
// 2(M) x 4(N) WMMA tiles of 16x16.  Double-buffered LDS filled by async
// global->LDS DMA; one barrier per k-step.  LDS row stride padded to 40
// elements (80B) for 16B alignment + bank spread.
// mode 0: store bf16 at [B,H,S,Dh]      (Q, K)
// mode 1: store bf16 at [B,H,Dh,S]      (V transposed)
// mode 2: store f32 row-major [M,N]     (final output)
#define LDA_PAD 40

__global__ __launch_bounds__(256) void gemm_bf16_wmma(
    const __bf16* __restrict__ A, const __bf16* __restrict__ W,
    const float* __restrict__ bias, void* __restrict__ out,
    int M, int N, int K, int mode) {
  __shared__ __bf16 sA[2][128 * LDA_PAD];
  __shared__ __bf16 sB[2][128 * LDA_PAD];

  const int tid  = threadIdx.x;
  const int lane = tid & 31;
  const int w    = tid >> 5;
  const int wm   = w & 3;        // 0..3 -> M offset 32*wm
  const int wn   = w >> 2;       // 0..1 -> N offset 64*wn
  const int half = lane >> 4;    // 0/1
  const int l16  = lane & 15;
  const int e0   = half * 8;

  const int m0 = blockIdx.x * 128;
  const int n0 = blockIdx.y * 128;

  const int r  = tid >> 2;        // 0..63
  const int c8 = (tid & 3) * 8;   // 0,8,16,24

  v8f acc[2][4] = {};

  // stage one 128x32 A tile + 128x32 B tile into LDS buffer `dst`
  auto stage = [&](int k0, __bf16* dA, __bf16* dB) {
#pragma unroll
    for (int p = 0; p < 2; ++p) {
      int row = r + p * 64;
      async_cp16(&A[(size_t)(m0 + row) * K + k0 + c8], &dA[row * LDA_PAD + c8]);
      async_cp16(&W[(size_t)(n0 + row) * K + k0 + c8], &dB[row * LDA_PAD + c8]);
    }
  };

  stage(0, sA[0], sB[0]);
  int buf = 0;

  for (int k0 = 0; k0 < K; k0 += 32) {
    wait_async_all();       // DMA into sA[buf]/sB[buf] complete (this wave)
    __syncthreads();        // all waves done filling buf / done reading buf^1
    if (k0 + 32 < K) stage(k0 + 32, sA[buf ^ 1], sB[buf ^ 1]);

    const __bf16* cA = sA[buf];
    const __bf16* cB = sB[buf];

    AFrag af[2];
#pragma unroll
    for (int mt = 0; mt < 2; ++mt) {
      int arow = wm * 32 + mt * 16 + l16;
      af[mt].h[0] = *(const v8bf*)&cA[arow * LDA_PAD + e0];
      af[mt].h[1] = *(const v8bf*)&cA[arow * LDA_PAD + e0 + 16];
    }
    AFrag bf[4];
#pragma unroll
    for (int nt = 0; nt < 4; ++nt) {
      int brow = wn * 64 + nt * 16 + l16;
      bf[nt].h[0] = *(const v8bf*)&cB[brow * LDA_PAD + e0];
      bf[nt].h[1] = *(const v8bf*)&cB[brow * LDA_PAD + e0 + 16];
    }
#pragma unroll
    for (int mt = 0; mt < 2; ++mt)
#pragma unroll
      for (int nt = 0; nt < 4; ++nt)
        acc[mt][nt] = wmma_bf16(af[mt].v, bf[nt].v, acc[mt][nt]);
    // fragment ds_loads are consumed by WMMA above, so DScnt has drained
    // before any wave reaches the next barrier -> buf^1 overwrite is safe.
    buf ^= 1;
  }

  // bias + store.  C layout: VGPR g, lane -> row = tile_m + g + 8*half,
  // col = tile_n + l16.
#pragma unroll
  for (int nt = 0; nt < 4; ++nt) {
    int n  = n0 + wn * 64 + nt * 16 + l16;
    float bv = bias[n];
#pragma unroll
    for (int mt = 0; mt < 2; ++mt) {
      int mbase = m0 + wm * 32 + mt * 16 + 8 * half;
#pragma unroll
      for (int g = 0; g < 8; ++g) {
        float v = acc[mt][nt][g] + bv;
        int m = mbase + g;
        if (mode == 2) {
          ((float*)out)[(size_t)m * N + n] = v;
        } else {
          int b = m >> 11;      // m / SEQ
          int s = m & (SEQ_ - 1);
          int h = n >> 6;       // n / DH
          int d = n & (DH_ - 1);
          size_t idx;
          if (mode == 0)
            idx = (((size_t)(b * NH_ + h)) * SEQ_ + s) * DH_ + d;
          else
            idx = (((size_t)(b * NH_ + h)) * DH_ + d) * SEQ_ + s;
          ((__bf16*)out)[idx] = (__bf16)v;
        }
      }
    }
  }
}

// ---------------------------------------------------------------- flash attention
// One wave per 16-query tile; 8 waves (128 queries) per block; grid (S/128, B*H).
// Q,K: [BH][S][64] bf16.  Vt: [BH][64][S] bf16.  O: [B][S][1024] bf16.
#define PLDS_STRIDE 40

__global__ __launch_bounds__(256) void attn_flash_wmma(
    const __bf16* __restrict__ Q, const __bf16* __restrict__ Kx,
    const __bf16* __restrict__ Vt, __bf16* __restrict__ O) {
  __shared__ __bf16 Pl[8][16 * PLDS_STRIDE];   // per-wave P bounce buffer

  const int tid  = threadIdx.x;
  const int lane = tid & 31;
  const int w    = tid >> 5;
  const int half = lane >> 4;
  const int l16  = lane & 15;
  const int e0   = half * 8;

  const int bh = blockIdx.y;                        // 0..63
  const int q0 = (blockIdx.x * 8 + w) * 16;         // query tile base
  const size_t base = (size_t)bh * SEQ_ * DH_;
  const __bf16* Qp = Q + base;
  const __bf16* Kp = Kx + base;
  const __bf16* Vp = Vt + base;

  // Q A-fragments for d 0..31 and 32..63 (held in registers all iterations)
  AFrag qa[2];
  {
    const __bf16* qr = Qp + (size_t)(q0 + l16) * DH_;
    qa[0].h[0] = *(const v8bf*)(qr + half * 16);
    qa[0].h[1] = *(const v8bf*)(qr + half * 16 + 8);
    qa[1].h[0] = *(const v8bf*)(qr + 32 + half * 16);
    qa[1].h[1] = *(const v8bf*)(qr + 32 + half * 16 + 8);
  }

  v8f o[4] = {};
  float mrow[8], lrow[8];
#pragma unroll
  for (int g = 0; g < 8; ++g) { mrow[g] = -3.0e38f; lrow[g] = 0.0f; }

  const float L2E = 1.4426950408889634f;
  const int kc0  = l16;                 // relative key col, tile 0
  const int qb   = q0 + 8 * half;       // this lane's row base

  for (int kb = 0; kb <= q0; kb += 32) {
    // ---- scores: two 16-key tiles, contract over d (two 32-wide WMMA steps)
    v8f s[2] = {};
#pragma unroll
    for (int t = 0; t < 2; ++t) {
      const __bf16* kr = Kp + (size_t)(kb + t * 16 + l16) * DH_ + half * 16;
      v16bf k0f = *(const v16bf*)(kr);        // d 0..31  (this lane's half)
      v16bf k1f = *(const v16bf*)(kr + 32);   // d 32..63
      s[t] = wmma_bf16(qa[0].v, k0f, s[t]);
      s[t] = wmma_bf16(qa[1].v, k1f, s[t]);
    }

    // ---- online softmax (rows live across 16 lanes of each half-wave)
    float mnew[8], fac[8];
#pragma unroll
    for (int g = 0; g < 8; ++g) {
      float a0 = s[0][g] * 0.125f;            // 1/sqrt(64)
      float a1 = s[1][g] * 0.125f;
      if (kb + kc0 > qb + g)       a0 = -3.0e38f;   // causal
      if (kb + kc0 + 16 > qb + g)  a1 = -3.0e38f;
      s[0][g] = a0; s[1][g] = a1;
      float mx = fmaxf(a0, a1);
      mx = fmaxf(mx, __shfl_xor(mx, 1));
      mx = fmaxf(mx, __shfl_xor(mx, 2));
      mx = fmaxf(mx, __shfl_xor(mx, 4));
      mx = fmaxf(mx, __shfl_xor(mx, 8));
      mnew[g] = fmaxf(mrow[g], mx);
      fac[g]  = __builtin_amdgcn_exp2f((mrow[g] - mnew[g]) * L2E);
      mrow[g] = mnew[g];
    }
#pragma unroll
    for (int g = 0; g < 8; ++g) {
      float p0 = __builtin_amdgcn_exp2f((s[0][g] - mnew[g]) * L2E);
      float p1 = __builtin_amdgcn_exp2f((s[1][g] - mnew[g]) * L2E);
      float rs = p0 + p1;
      rs += __shfl_xor(rs, 1);
      rs += __shfl_xor(rs, 2);
      rs += __shfl_xor(rs, 4);
      rs += __shfl_xor(rs, 8);
      lrow[g] = lrow[g] * fac[g] + rs;
      // stash P in LDS (C layout -> row-major 16x32)
      int row = g + 8 * half;
      Pl[w][row * PLDS_STRIDE + l16]      = (__bf16)p0;
      Pl[w][row * PLDS_STRIDE + 16 + l16] = (__bf16)p1;
    }
    // rescale running output
    v8f fv;
#pragma unroll
    for (int g = 0; g < 8; ++g) fv[g] = fac[g];
#pragma unroll
    for (int t = 0; t < 4; ++t) o[t] *= fv;

    // ---- P @ V : re-read P in A layout (DS ops are in-order per wave)
    AFrag pf;
    pf.h[0] = *(const v8bf*)&Pl[w][l16 * PLDS_STRIDE + half * 16];
    pf.h[1] = *(const v8bf*)&Pl[w][l16 * PLDS_STRIDE + half * 16 + 8];
#pragma unroll
    for (int t = 0; t < 4; ++t) {
      // Vt[d][key]: lane holds column d = t*16 + l16, keys kb+half*16 .. +15
      v16bf vf = *(const v16bf*)(Vp + (size_t)(t * 16 + l16) * SEQ_ + kb + half * 16);
      o[t] = wmma_bf16(pf.v, vf, o[t]);
    }
  }

  // ---- normalize + store to [B][S][D] bf16
  const int b = bh >> 4, h = bh & 15;
#pragma unroll
  for (int g = 0; g < 8; ++g) {
    float inv = 1.0f / lrow[g];
    int srow = q0 + g + 8 * half;
    size_t rowbase = ((size_t)(b * SEQ_ + srow)) * DM_ + h * DH_;
#pragma unroll
    for (int t = 0; t < 4; ++t)
      O[rowbase + t * 16 + l16] = (__bf16)(o[t][g] * inv);
  }
}

// ---------------------------------------------------------------- launch
extern "C" void kernel_launch(void* const* d_in, const int* in_sizes, int n_in,
                              void* d_out, int out_size, void* d_ws, size_t ws_size,
                              hipStream_t stream) {
  (void)in_sizes; (void)n_in; (void)out_size; (void)ws_size;
  const float* x  = (const float*)d_in[0];
  // d_in[1] = boolean causal mask: implemented analytically in the kernel
  const float* Wq = (const float*)d_in[2];
  const float* bq = (const float*)d_in[3];
  const float* Wk = (const float*)d_in[4];
  const float* bk = (const float*)d_in[5];
  const float* Wv = (const float*)d_in[6];
  const float* bv = (const float*)d_in[7];
  const float* Wo = (const float*)d_in[8];
  const float* bo = (const float*)d_in[9];

  char* ws = (char*)d_ws;
  const size_t MB = 1u << 20;
  __bf16* xb  = (__bf16*)(ws + 0);         // 16 MB  (reused as AOb later)
  __bf16* wqb = (__bf16*)(ws + 16 * MB);   //  2 MB
  __bf16* wkb = (__bf16*)(ws + 18 * MB);
  __bf16* wvb = (__bf16*)(ws + 20 * MB);
  __bf16* wob = (__bf16*)(ws + 22 * MB);
  __bf16* Qb  = (__bf16*)(ws + 24 * MB);   // 16 MB  [B,H,S,Dh]
  __bf16* Kb  = (__bf16*)(ws + 40 * MB);   // 16 MB  [B,H,S,Dh]
  __bf16* Vtb = (__bf16*)(ws + 56 * MB);   // 16 MB  [B,H,Dh,S]
  __bf16* AOb = (__bf16*)(ws + 0);         // attention out bf16, overlays xb

  const int NX = MTOT_ * DM_;   // 8388608
  const int NW = DM_ * DM_;     // 1048576

  cvt_f32_bf16<<<NX / (256 * 4), 256, 0, stream>>>(x,  xb,  NX);
  cvt_f32_bf16<<<NW / (256 * 4), 256, 0, stream>>>(Wq, wqb, NW);
  cvt_f32_bf16<<<NW / (256 * 4), 256, 0, stream>>>(Wk, wkb, NW);
  cvt_f32_bf16<<<NW / (256 * 4), 256, 0, stream>>>(Wv, wvb, NW);
  cvt_f32_bf16<<<NW / (256 * 4), 256, 0, stream>>>(Wo, wob, NW);

  dim3 gg(MTOT_ / 128, DM_ / 128);   // (64, 8)
  gemm_bf16_wmma<<<gg, 256, 0, stream>>>(xb, wqb, bq, Qb,  MTOT_, DM_, DM_, 0);
  gemm_bf16_wmma<<<gg, 256, 0, stream>>>(xb, wkb, bk, Kb,  MTOT_, DM_, DM_, 0);
  gemm_bf16_wmma<<<gg, 256, 0, stream>>>(xb, wvb, bv, Vtb, MTOT_, DM_, DM_, 1);

  attn_flash_wmma<<<dim3(SEQ_ / 128, 4 * NH_), 256, 0, stream>>>(Qb, Kb, Vtb, AOb);

  gemm_bf16_wmma<<<gg, 256, 0, stream>>>(AOb, wob, bo, d_out, MTOT_, DM_, DM_, 2);
}
